// ActorNetwork_12781822672981
// MI455X (gfx1250) — matrix operand order repaired and checked by reading, not compiled, not run on hardware
//
#include <hip/hip_runtime.h>
#include <hip/hip_bf16.h>
#include <math.h>

typedef __attribute__((ext_vector_type(16))) _Float16 v16h;
typedef __attribute__((ext_vector_type(8)))  _Float16 v8h;
typedef __attribute__((ext_vector_type(4)))  _Float16 v4h;
typedef __attribute__((ext_vector_type(8)))  float    v8f;

#define B_TOTAL 32768
#define DIM     376
#define G_NUM   8
#define H1      400
#define H2      300
#define A_DIM   17

#define K1_BLK  12          // ceil(376/32)
#define K2_BLK  13          // ceil(400/32) -> 416
#define K3_BLK  10          // ceil(300/32) -> 320
#define NT1     25          // 400/16
#define NT2     19          // ceil(300/16)
#define NT3     2           // ceil(17/16)
#define D_PAD   384
#define H1_PAD  416
#define H2_PAD  320

// swizzled-weight sizes (halves); tile (nt,kb) = 512 halves, nt-major so each
// n-tile's K-strip is one contiguous KB*1KB block (bulk async-copy friendly).
#define SW1_PER_G (NT1 * K1_BLK * 512)
#define SW2_TOT   (NT2 * K2_BLK * 512)
#define SW3_PER_G (NT3 * K3_BLK * 512)
#define SW1_TOT   (G_NUM * SW1_PER_G)
#define SW3_TOT   (G_NUM * SW3_PER_G)

#define BBUF_HALVES (K2_BLK * 512)          // largest strip: 13 KB
#define SMEM_HALVES (64 * H1_PAD + 64 * D_PAD + 2 * BBUF_HALVES)

// Scheduling-template hooks (LLVM sched_group_barrier; masks: 0x100 = DS read,
// 0x008 = MFMA/WMMA). No-ops if the builtin is unavailable.
#if defined(__has_builtin)
#if __has_builtin(__builtin_amdgcn_sched_group_barrier)
#define SG_DS(n)   __builtin_amdgcn_sched_group_barrier(0x100, (n), 0)
#define SG_WMMA(n) __builtin_amdgcn_sched_group_barrier(0x008, (n), 0)
#endif
#endif
#ifndef SG_DS
#define SG_DS(n)
#define SG_WMMA(n)
#endif

// ---------------------------------------------------------------------------
// Setup kernel: f32 weights -> f16 in WMMA B-fragment order, nt-major strips.
// b[h] for lane l of tile (nt,kb): k = kb*32 + (l>>4)*16 + h ; n = nt*16 + (l&15)
// K/N out-of-range baked in as zeros -> no edge masks in the GEMM loops.
// ---------------------------------------------------------------------------
__device__ __forceinline__ _Float16 swz_elem(const float* __restrict__ src,
                                             int K, int N, int KB, int i) {
    int h    = i & 15;
    int lane = (i >> 4) & 31;
    int tile = i >> 9;               // nt*KB + kb
    int kb   = tile % KB;
    int nt   = tile / KB;
    int k    = kb * 32 + (lane >> 4) * 16 + h;
    int n    = nt * 16 + (lane & 15);
    float v  = (k < K && n < N) ? src[(size_t)k * N + n] : 0.0f;
    return (_Float16)v;
}

__global__ void cvt_weights(const float* __restrict__ W1,
                            const float* __restrict__ W2,
                            const float* __restrict__ W3,
                            _Float16* __restrict__ dst) {
    const int total = SW1_TOT + SW2_TOT + SW3_TOT;
    for (int i = blockIdx.x * blockDim.x + threadIdx.x; i < total;
         i += gridDim.x * blockDim.x) {
        _Float16 v;
        if (i < SW1_TOT) {
            int g = i / SW1_PER_G, loc = i % SW1_PER_G;
            v = swz_elem(W1 + (size_t)g * DIM * H1, DIM, H1, K1_BLK, loc);
        } else if (i < SW1_TOT + SW2_TOT) {
            v = swz_elem(W2, H1, H2, K2_BLK, i - SW1_TOT);
        } else {
            int j = i - SW1_TOT - SW2_TOT;
            int g = j / SW3_PER_G, loc = j % SW3_PER_G;
            v = swz_elem(W3 + (size_t)g * H2 * A_DIM, H2, A_DIM, K3_BLK, loc);
        }
        dst[i] = v;
    }
}

// ---------------------------------------------------------------------------
// CDNA5 async global->LDS bulk copy of one weight K-strip (ASYNCcnt-tracked).
// ---------------------------------------------------------------------------
__device__ __forceinline__ void stage_b_async(const _Float16* __restrict__ src,
                                              _Float16* dst, int bytes, int tid) {
    unsigned ldsbase = (unsigned)(size_t)dst;       // LDS byte offset (AS3)
    const char* s = (const char*)src;
    for (int i = tid * 16; i < bytes; i += 128 * 16) {
        unsigned ldsa = ldsbase + (unsigned)i;
        unsigned long long ga = (unsigned long long)(s + i);
        asm volatile("global_load_async_to_lds_b128 %0, %1, off"
                     :: "v"(ldsa), "v"(ga) : "memory");
    }
}

__device__ __forceinline__ void wait_async() {
    asm volatile("s_wait_asynccnt 0x0" ::: "memory");
}

// ---------------------------------------------------------------------------
// Fragment loaders
// ---------------------------------------------------------------------------
__device__ __forceinline__ v16h load_a_lds(const _Float16* base, int ld,
                                           int row, int hi, int k0) {
    const _Float16* p = base + row * ld + k0 + hi * 16;
    v8h lo = *(const v8h*)(p);
    v8h hh = *(const v8h*)(p + 8);
    return __builtin_shufflevector(lo, hh, 0, 1, 2, 3, 4, 5, 6, 7,
                                   8, 9, 10, 11, 12, 13, 14, 15);
}

__device__ __forceinline__ v16h load_b_frag(const _Float16* bb, int kb, int lane) {
    return *(const v16h*)(bb + (kb << 9) + (lane << 4));
}

#define WMMA_F16(A, B, C) \
    __builtin_amdgcn_wmma_f32_16x16x32_f16(false, (A), false, (B), (short)0, (C), false, false)

// ---------------------------------------------------------------------------
// One GEMM layer: A fragments in registers, B strips double-buffered in LDS
// via async copy (shared by all 4 waves), K-accumulation split into 3 WMMA
// chains, and an explicit sched_group_barrier template that pins the pipeline
//   [DS x6][WMMA][DS x2][WMMA][DS x2][WMMA][DS x2]...
// so every WMMA runs with >=2 fragment-load pairs still in flight.
// ---------------------------------------------------------------------------
template <int KB, int NT, typename Epi>
__device__ __forceinline__ void run_layer(const _Float16* __restrict__ Wg,
                                          const _Float16* sA_wave, int lda,
                                          _Float16* s_b, int tid, int lane,
                                          int row, int hi, Epi epi) {
    constexpr int H  = (KB + 2) / 3;             // chain-0/1 length (ceil(KB/3))
    constexpr int L2 = KB - 2 * H;               // chain-2 length

    v16h afr[KB];
#pragma unroll
    for (int kb = 0; kb < KB; ++kb)
        afr[kb] = load_a_lds(sA_wave, lda, row, hi, kb * 32);

    stage_b_async(Wg, s_b, KB * 1024, tid);
    wait_async();
    __syncthreads();
    for (int nt = 0; nt < NT; ++nt) {
        const _Float16* bb = s_b + (nt & 1) * BBUF_HALVES;
        if (nt + 1 < NT)                                   // prefetch next strip
            stage_b_async(Wg + (size_t)(nt + 1) * KB * 512,
                          s_b + ((nt + 1) & 1) * BBUF_HALVES, KB * 1024, tid);
        v8f c0 = {}, c1 = {}, c2 = {};
#pragma unroll
        for (int j = 0; j < H; ++j) {
            {
                v16h b = load_b_frag(bb, j, lane);
                c0 = WMMA_F16(afr[j], b, c0);
            }
            if (j + H < KB) {
                v16h b = load_b_frag(bb, j + H, lane);
                c1 = WMMA_F16(afr[j + H], b, c1);
            }
            if (j + 2 * H < KB) {
                v16h b = load_b_frag(bb, j + 2 * H, lane);
                c2 = WMMA_F16(afr[j + 2 * H], b, c2);
            }
        }
        // ---- scheduling template: heads of all 3 chains load first, then
        // each WMMA is followed by the next fragment's 2 ds reads ----------
        SG_DS(6);
#pragma unroll
        for (int j = 0; j < H; ++j) {
            SG_WMMA(1);                          // chain 0, frag j
            if (j + 1 < H) SG_DS(2);
            if (j + H < KB) {
                SG_WMMA(1);                      // chain 1, frag j+H
                if (j + 1 < H) SG_DS(2);
            }
            if (j + 2 * H < KB) {
                SG_WMMA(1);                      // chain 2, frag j+2H
                if (j + 1 < L2) SG_DS(2);
            }
        }
        v8f c;
#pragma unroll
        for (int i = 0; i < 8; ++i)
            c[i] = (c0[i] + c1[i]) + c2[i];
        epi(nt, c);
        wait_async();                                      // staging landed
        __syncthreads();                                   // publish + reuse ok
    }
}

// ---------------------------------------------------------------------------
// Fused actor network: 128 threads (4 wave32) per 64-row tile; every weight
// fragment fetched once per 64 rows (4x less L2 traffic than per-wave).
// ---------------------------------------------------------------------------
__global__ void __launch_bounds__(128)
actor_fused(const float* __restrict__ state,
            const long long* __restrict__ idx,
            const float* __restrict__ b1,
            const float* __restrict__ b2,
            const float* __restrict__ b3,
            const _Float16* __restrict__ W1sw,
            const _Float16* __restrict__ W2sw,
            const _Float16* __restrict__ W3sw,
            float* __restrict__ out) {
    __shared__ _Float16 smem[SMEM_HALVES];
    _Float16* s_h1 = smem;                       // 64 x 416
    _Float16* s_xf = smem + 64 * H1_PAD;         // union: x (64x384) / h_f (64x320)
    _Float16* s_b  = smem + 64 * H1_PAD + 64 * D_PAD;   // 2 x 13KB strips

    const int tid  = threadIdx.x;
    const int lane = tid & 31;
    const int wv   = tid >> 5;                   // wave id 0..3
    const int row  = lane & 15;
    const int hi   = lane >> 4;
    const int row0 = blockIdx.x * 64;
    const int wrow = wv * 16;                    // this wave's row base in tile

    // zero h1 K-pad columns once (never overwritten afterwards)
    for (int i = tid; i < 64 * (H1_PAD - H1); i += 128) {
        int r = i / (H1_PAD - H1), c = i % (H1_PAD - H1);
        s_h1[r * H1_PAD + H1 + c] = (_Float16)0.0f;
    }

    const int g0 = (int)idx[row0];
    const int g1 = (int)idx[row0 + 63];          // idx is sorted

    for (int g = g0; g <= g1; ++g) {             // 1 iteration except seg edges
        // ---- stage x tile (f32 -> f16, zero K-pad); h_f overlay => per g ----
        for (int i = tid; i < 64 * (DIM / 4); i += 128) {
            int r = i / (DIM / 4), c4 = (i % (DIM / 4)) * 4;
            float4 v = *(const float4*)(state + (size_t)(row0 + r) * DIM + c4);
            v4h h4 = { (_Float16)v.x, (_Float16)v.y, (_Float16)v.z, (_Float16)v.w };
            *(v4h*)(s_xf + r * D_PAD + c4) = h4;
        }
        for (int i = tid; i < 64 * (D_PAD - DIM); i += 128) {
            int r = i / (D_PAD - DIM), c = i % (D_PAD - DIM);
            s_xf[r * D_PAD + DIM + c] = (_Float16)0.0f;
        }
        __syncthreads();

        // ================= layer 1: h1 = relu(x @ W1[g] + b1[g]) ===========
        {
            const _Float16* Wg = W1sw + (size_t)g * SW1_PER_G;
            const float*    bg = b1 + g * H1;
            run_layer<K1_BLK, NT1>(Wg, s_xf + wrow * D_PAD, D_PAD, s_b,
                                   tid, lane, row, hi,
                [&](int nt, v8f c) {
                    int n = nt * 16 + row;
                    float bias = bg[n];
#pragma unroll
                    for (int r8 = 0; r8 < 8; ++r8) {
                        float v = c[r8] + bias; v = v > 0.0f ? v : 0.0f;
                        s_h1[(wrow + r8 + 8 * hi) * H1_PAD + n] = (_Float16)v;
                    }
                });
        }
        // ================= layer 2: h_f = relu(h1 @ W2 + b2) ===============
        run_layer<K2_BLK, NT2>(W2sw, s_h1 + wrow * H1_PAD, H1_PAD, s_b,
                               tid, lane, row, hi,
            [&](int nt, v8f c) {
                int n = nt * 16 + row;
                if (n < H2) {
                    float bias = b2[n];
#pragma unroll
                    for (int r8 = 0; r8 < 8; ++r8) {
                        float v = c[r8] + bias; v = v > 0.0f ? v : 0.0f;
                        s_xf[(wrow + r8 + 8 * hi) * H2_PAD + n] = (_Float16)v;
                    }
                }
            });
        // re-zero h_f K-pad columns (overlay held x data there)
        for (int i = tid; i < 64 * (H2_PAD - H2); i += 128) {
            int r = i / (H2_PAD - H2), c = i % (H2_PAD - H2);
            s_xf[r * H2_PAD + H2 + c] = (_Float16)0.0f;
        }
        __syncthreads();
        // ========= layer 3: out = tanh(h_f @ W3[g] + b3[g]), masked ========
        {
            const _Float16* Wg = W3sw + (size_t)g * SW3_PER_G;
            const float*    bg = b3 + g * A_DIM;
            bool ok[8];
#pragma unroll
            for (int r8 = 0; r8 < 8; ++r8)
                ok[r8] = ((int)idx[row0 + wrow + r8 + 8 * hi] == g);
            run_layer<K3_BLK, NT3>(Wg, s_xf + wrow * H2_PAD, H2_PAD, s_b,
                                   tid, lane, row, hi,
                [&](int nt, v8f c) {
                    int n = nt * 16 + row;
                    if (n < A_DIM) {
                        float bias = bg[n];
#pragma unroll
                        for (int r8 = 0; r8 < 8; ++r8) {
                            if (ok[r8])
                                out[(size_t)(row0 + wrow + r8 + 8 * hi) * A_DIM + n]
                                    = tanhf(c[r8] + bias);
                        }
                    }
                });
        }
    }
}

// ---------------------------------------------------------------------------
extern "C" void kernel_launch(void* const* d_in, const int* in_sizes, int n_in,
                              void* d_out, int out_size, void* d_ws, size_t ws_size,
                              hipStream_t stream) {
    const float*     state = (const float*)d_in[0];
    const long long* idx   = (const long long*)d_in[1];
    const float*     W1    = (const float*)d_in[2];
    const float*     b1    = (const float*)d_in[3];
    const float*     W2    = (const float*)d_in[4];
    const float*     b2    = (const float*)d_in[5];
    const float*     W3    = (const float*)d_in[6];
    const float*     b3    = (const float*)d_in[7];

    _Float16* wsh  = (_Float16*)d_ws;
    _Float16* W1sw = wsh;
    _Float16* W2sw = wsh + SW1_TOT;
    _Float16* W3sw = wsh + SW1_TOT + SW2_TOT;

    const int total = SW1_TOT + SW2_TOT + SW3_TOT;   // ~1.44M halves, 2.9 MB
    cvt_weights<<<(total + 255) / 256, 256, 0, stream>>>(W1, W2, W3, wsh);

    actor_fused<<<B_TOTAL / 64, 128, 0, stream>>>(state, idx, b1, b2, b3,
                                                  W1sw, W2sw, W3sw,
                                                  (float*)d_out);
}
